// HierarchicalPolicy_30331059044712
// MI455X (gfx1250) — matrix operand order repaired and checked by reading, not compiled, not run on hardware
//
#include <hip/hip_runtime.h>
#include <hip/hip_bf16.h>

typedef __attribute__((ext_vector_type(16))) _Float16 v16h;
typedef __attribute__((ext_vector_type(8)))  float    v8f;
typedef __attribute__((ext_vector_type(4)))  unsigned int u32x4;
typedef __attribute__((ext_vector_type(8)))  unsigned int u32x8;

#define NEGV (-1000000000.0f)

__device__ inline v8f wmma32f16(v16h a, v16h b, v8f c) {
  return __builtin_amdgcn_wmma_f32_16x16x32_f16(false, a, false, b, (short)0, c, false, false);
}

__device__ inline void atomAddF(float* p, float v) { unsafeAtomicAdd(p, v); }

union HFrag { v16h v; _Float16 h[16]; };

// A fragment (16x32 f16): lanes 0-15 row=lane, K = kb+{0..15};
// lanes 16-31 row=lane-16, K = kb+{16..31}.
__device__ inline v16h load_a_frag(const _Float16* lds, int ldk, int kb, int lane) {
  int row  = lane & 15;
  int koff = kb + ((lane & 16) ? 16 : 0);
  const _Float16* p = lds + row * ldk + koff;
  HFrag f;
#pragma unroll
  for (int i = 0; i < 16; i++) f.h[i] = p[i];
  return f.v;
}

// B fragment (32x16 f16): lane n<16 -> col n, K=kb..kb+15; lane n>=16 -> col n-16, K=kb+16..31.
// Bt stored transposed [C][ldk] so each lane reads 16 contiguous halves.
__device__ inline v16h load_b_frag(const _Float16* Bt, int ldk, int col0, int kb, int lane) {
  int col   = col0 + (lane & 15);
  int kbase = kb + ((lane & 16) ? 16 : 0);
  const _Float16* p = Bt + (long)col * ldk + kbase;
  HFrag f;
#pragma unroll
  for (int i = 0; i < 16; i++) f.h[i] = p[i];
  return f.v;
}

// Tensor Data Mover: DMA a 16-row x 128-col f16 tile (row-major, dense) from global
// into LDS at byte offset lds_off. 2D D# per CDNA5 ISA:
//   tensor = rows_total x 128 elems, data_size=2B, tile = 16 x 128, OOB rows read as 0.
__device__ inline void tdm_load_tile_f16(unsigned lds_off, const void* gbase,
                                         unsigned rows_total, unsigned row0) {
  unsigned long long ga = (unsigned long long)gbase + (unsigned long long)row0 * 256ull;
  u32x4 g0 = { 1u,                                  // count=1 (valid user descriptor)
               lds_off,                              // lds_addr
               (unsigned)ga,                         // global_addr[31:0]
               ((unsigned)(ga >> 32) & 0x01FFFFFFu) | 0x80000000u }; // addr[56:32], type=2
  u32x8 g1 = { 0x00010000u,                          // wg_mask=0, data_size=1 (2 bytes)
               128u << 16,                           // tensor_dim0[15:0] = 128
               (rows_total & 0xFFFFu) << 16,         // tensor_dim1[15:0]
               (rows_total >> 16) | (128u << 16),    // tensor_dim1[31:16], tile_dim0=128
               16u,                                  // tile_dim1=16, tile_dim2=0
               128u,                                 // tensor_dim0_stride[31:0] = 128
               0u, 0u };                             // stride0 hi, dim1_stride (unused, 2D)
  asm volatile("tensor_load_to_lds %0, %1" :: "s"(g0), "s"(g1) : "memory");
}

__global__ void hp_zero_kernel(float* __restrict__ p, long n) {
  long i = (long)blockIdx.x * blockDim.x + threadIdx.x;
  if (i < n) p[i] = 0.f;
}

// Precompute f16 transposed weight buffers + collapsed edge-constant weights:
// Wc = W_ep @ W_msg[128:,:]  (transposed, zero-padded K 16->32)
// bias_vec = b_ep @ W_msg[128:,:] + b_msg
__global__ void hp_prep_kernel(const float* __restrict__ W_np, const float* __restrict__ W_ep,
                               const float* __restrict__ b_ep, const float* __restrict__ W_msg,
                               const float* __restrict__ b_msg, const float* __restrict__ W_ih,
                               const float* __restrict__ W_hh,
                               _Float16* __restrict__ Bnp, _Float16* __restrict__ Bc,
                               _Float16* __restrict__ Btm, _Float16* __restrict__ Bih,
                               _Float16* __restrict__ Bhh, float* __restrict__ bias_vec) {
  int tid = threadIdx.x;
  for (int idx = tid; idx < 128 * 32; idx += 256) {        // Bnp[c][k] = W_np[k][c]
    int c = idx >> 5, k = idx & 31;
    Bnp[idx] = (k < 16) ? (_Float16)W_np[k * 128 + c] : (_Float16)0.f;
  }
  for (int idx = tid; idx < 128 * 32; idx += 256) {        // Bc[c][j] = Wc[j][c]
    int c = idx >> 5, j = idx & 31;
    if (j < 16) {
      float s = 0.f;
      for (int k = 0; k < 128; k++) s += W_ep[j * 128 + k] * W_msg[(128 + k) * 128 + c];
      Bc[idx] = (_Float16)s;
    } else Bc[idx] = (_Float16)0.f;
  }
  for (int c = tid; c < 128; c += 256) {
    float s = b_msg[c];
    for (int k = 0; k < 128; k++) s += b_ep[k] * W_msg[(128 + k) * 128 + c];
    bias_vec[c] = s;
  }
  for (int idx = tid; idx < 128 * 128; idx += 256) {       // Btm[c][k] = W_msg_top[k][c]
    int c = idx >> 7, k = idx & 127;
    Btm[idx] = (_Float16)W_msg[k * 128 + c];
  }
  for (int idx = tid; idx < 384 * 128; idx += 256) {       // W_ih/W_hh already [out][k]
    Bih[idx] = (_Float16)W_ih[idx];
    Bhh[idx] = (_Float16)W_hh[idx];
  }
}

// Generic row-block GEMM: Out[M,C] = A[M,K] @ Bt^T + bias ; f16 WMMA, f32 accumulate.
// Optionally writes an f16 mirror of the output (consumed later by TDM-staged GEMMs).
__global__ void hp_gemm_kernel(const float* __restrict__ A, int M, int K, int Kpad,
                               const _Float16* __restrict__ Bt, const float* __restrict__ bias,
                               float* __restrict__ Out, _Float16* __restrict__ Out16, int C) {
  __shared__ _Float16 sA[16 * 128];
  int tid = threadIdx.x;
  long row0 = (long)blockIdx.x * 16;
  for (int idx = tid; idx < 16 * Kpad; idx += 256) {
    int r = idx / Kpad, k = idx % Kpad;
    long gr = row0 + r;
    float v = (gr < M && k < K) ? A[gr * (long)K + k] : 0.f;
    sA[idx] = (_Float16)v;
  }
  __syncthreads();
  int wave = tid >> 5, lane = tid & 31;
  for (int ct = wave; ct * 16 < C; ct += 8) {
    int col0 = ct * 16;
    v8f acc = {};
    for (int kb = 0; kb < Kpad; kb += 32) {
      v16h a = load_a_frag(sA, Kpad, kb, lane);
      v16h b = load_b_frag(Bt, Kpad, col0, kb, lane);
      acc = wmma32f16(a, b, acc);
    }
    int nn = lane & 15, mb = (lane & 16) ? 8 : 0, col = col0 + nn;
    float bv = bias ? bias[col] : 0.f;
#pragma unroll
    for (int r = 0; r < 8; r++) {
      long gr = row0 + mb + r;
      if (gr < M) {
        float o = acc[r] + bv;
        Out[gr * (long)C + col] = o;
        if (Out16) Out16[gr * (long)C + col] = (_Float16)o;
      }
    }
  }
}

// Per-edge constant part: se[dst] += edge_feat[e], cnt[dst] += 1  (once, reused all 3 iters)
__global__ void hp_seg_edge_kernel(const float* __restrict__ ef, const int* __restrict__ dst,
                                   long E, float* __restrict__ se, float* __restrict__ cnt) {
  long eid = (long)blockIdx.x * blockDim.x + threadIdx.x;
  if (eid >= E) return;
  long d = dst[eid];
  const float* p = ef + eid * 16;
  float* o = se + d * 16;
#pragma unroll
  for (int j = 0; j < 16; j++) atomAddF(o + j, p[j]);
  atomAddF(cnt + d, 1.f);
}

// hs[dst] += h[src] : 32 lanes/edge, float4 per lane; h is L2-resident (25.6MB << 192MB L2)
__global__ void hp_scatter_h_kernel(const float* __restrict__ h, const int* __restrict__ src,
                                    const int* __restrict__ dst, long E, float* __restrict__ hs) {
  long gid = (long)blockIdx.x * blockDim.x + threadIdx.x;
  long eid = gid >> 5;
  if (eid >= E) return;
  int c4 = (int)(gid & 31) << 2;
  long s = src[eid], d = dst[eid];
  const float4 v = *(const float4*)(h + s * 128 + c4);
  float* o = hs + d * 128 + c4;
  atomAddF(o + 0, v.x); atomAddF(o + 1, v.y); atomAddF(o + 2, v.z); atomAddF(o + 3, v.w);
}

// agg16 = f16( hs @ Wt + se @ Wc + cnt*bias_vec )  -- only the f16 mirror is needed (GRU/TDM input)
__global__ void hp_agg_kernel(const float* __restrict__ hs, const float* __restrict__ se,
                              const float* __restrict__ cnt, int N,
                              const _Float16* __restrict__ Btm, const _Float16* __restrict__ Bc,
                              const float* __restrict__ bias_vec, _Float16* __restrict__ agg16) {
  __shared__ _Float16 sA[16 * 128];
  __shared__ _Float16 sS[16 * 32];
  __shared__ float    sC[16];
  int tid = threadIdx.x;
  long row0 = (long)blockIdx.x * 16;
  for (int idx = tid; idx < 16 * 128; idx += 256) {
    int r = idx >> 7, k = idx & 127;
    long gr = row0 + r;
    sA[idx] = (gr < N) ? (_Float16)hs[gr * 128 + k] : (_Float16)0.f;
  }
  for (int idx = tid; idx < 16 * 32; idx += 256) {
    int r = idx >> 5, k = idx & 31;
    long gr = row0 + r;
    sS[idx] = (gr < N && k < 16) ? (_Float16)se[gr * 16 + k] : (_Float16)0.f;
  }
  if (tid < 16) { long gr = row0 + tid; sC[tid] = (gr < N) ? cnt[gr] : 0.f; }
  __syncthreads();
  int wave = tid >> 5, lane = tid & 31;
  int col0 = wave * 16;
  v8f acc = {};
  for (int kb = 0; kb < 128; kb += 32)
    acc = wmma32f16(load_a_frag(sA, 128, kb, lane), load_b_frag(Btm, 128, col0, kb, lane), acc);
  acc = wmma32f16(load_a_frag(sS, 32, 0, lane), load_b_frag(Bc, 32, col0, 0, lane), acc);
  int nn = lane & 15, mb = (lane & 16) ? 8 : 0, col = col0 + nn;
  float bv = bias_vec[col];
#pragma unroll
  for (int r = 0; r < 8; r++) {
    long gr = row0 + mb + r;
    if (gr < N) agg16[gr * 128 + col] = (_Float16)(acc[r] + sC[mb + r] * bv);
  }
}

// Fused GRU. A-tiles (agg16, h16) are DMA'd into LDS by the Tensor Data Mover;
// gate blend happens directly on WMMA accumulator fragments; h (f32) + h16 updated in place.
__global__ void hp_gru_kernel(const _Float16* __restrict__ agg16, float* __restrict__ h,
                              _Float16* __restrict__ h16, int N,
                              const _Float16* __restrict__ Bih, const _Float16* __restrict__ Bhh,
                              const float* __restrict__ b_ih, const float* __restrict__ b_hh) {
  extern __shared__ _Float16 smem[];           // [0,4096) agg tile, [4096,8192) h tile
  _Float16* sAgg = smem;
  _Float16* sH   = smem + 16 * 128;
  int tid = threadIdx.x;
  long row0 = (long)blockIdx.x * 16;
  if (tid < 32) {                              // wave 0 issues both TDM descriptors
    tdm_load_tile_f16(0u,    agg16, (unsigned)N, (unsigned)row0);
    tdm_load_tile_f16(4096u, h16,   (unsigned)N, (unsigned)row0);
    __builtin_amdgcn_s_wait_tensorcnt(0);
  }
  __syncthreads();
  int wave = tid >> 5, lane = tid & 31;
  int colA = wave * 16;                        // wave w owns gate columns colA..colA+15 of r,z,n
  v8f ir = {}, iz = {}, in_ = {}, hr = {}, hz = {}, hn = {};
  for (int kb = 0; kb < 128; kb += 32) {
    v16h aA = load_a_frag(sAgg, 128, kb, lane);
    v16h aH = load_a_frag(sH,   128, kb, lane);
    ir  = wmma32f16(aA, load_b_frag(Bih, 128, colA,       kb, lane), ir);
    iz  = wmma32f16(aA, load_b_frag(Bih, 128, 128 + colA, kb, lane), iz);
    in_ = wmma32f16(aA, load_b_frag(Bih, 128, 256 + colA, kb, lane), in_);
    hr  = wmma32f16(aH, load_b_frag(Bhh, 128, colA,       kb, lane), hr);
    hz  = wmma32f16(aH, load_b_frag(Bhh, 128, 128 + colA, kb, lane), hz);
    hn  = wmma32f16(aH, load_b_frag(Bhh, 128, 256 + colA, kb, lane), hn);
  }
  int nn = lane & 15, mb = (lane & 16) ? 8 : 0, col = colA + nn;
  float bir = b_ih[col], biz = b_ih[128 + col], bin = b_ih[256 + col];
  float bhr = b_hh[col], bhz = b_hh[128 + col], bhn = b_hh[256 + col];
#pragma unroll
  for (int r = 0; r < 8; r++) {
    long gr = row0 + mb + r;
    if (gr < N) {
      float hold = h[gr * 128 + col];
      float rr = 1.f / (1.f + expf(-(ir[r] + bir + hr[r] + bhr)));
      float zz = 1.f / (1.f + expf(-(iz[r] + biz + hz[r] + bhz)));
      float nv = tanhf(in_[r] + bin + rr * (hn[r] + bhn));
      float hnew = (1.f - zz) * nv + zz * hold;
      h[gr * 128 + col]   = hnew;
      h16[gr * 128 + col] = (_Float16)hnew;
    }
  }
}

__global__ void hp_colsum_kernel(const float* __restrict__ h, int N, float* __restrict__ g) {
  int c = threadIdx.x;
  float s = 0.f;
  for (long r = blockIdx.x; r < N; r += gridDim.x) s += h[r * 128 + c];
  atomAddF(&g[c], s);
}

// All three agents + value head, single block (tiny).
__global__ void hp_heads_kernel(const float* __restrict__ edge_feat, const float* __restrict__ W_ep,
                                const float* __restrict__ b_ep, const int* __restrict__ paths,
                                const int* __restrict__ path_mask, const float* __restrict__ path_features,
                                const int* __restrict__ mod_masks, const int* __restrict__ spec_masks,
                                const float* __restrict__ path_spectrum, const float* __restrict__ W_path,
                                const float* __restrict__ b_path, const float* __restrict__ W_mod,
                                const float* __restrict__ b_mod, const float* __restrict__ c1w,
                                const float* __restrict__ c1b, const float* __restrict__ c2w,
                                const float* __restrict__ c2b, const float* __restrict__ W_val,
                                const float* __restrict__ b_val, const float* __restrict__ g,
                                int N, int P, float* __restrict__ out) {
  __shared__ float z[16 * 128];
  __shared__ float lps[16];
  __shared__ float lm[6];
  __shared__ float xm[131];
  __shared__ float sp[324];
  __shared__ float lf[320];
  __shared__ int selP, selM;
  __shared__ float logpP, logpM;
  int tid = threadIdx.x;
  for (int idx = tid; idx < P * 128; idx += 320) {       // z[p] = mean_l(e[paths[p,l]])
    int p = idx >> 7, c = idx & 127;
    float s = 0.f;
    for (int l = 0; l < 8; l++) {
      const float* ef = edge_feat + (long)paths[p * 8 + l] * 16;
      float a = 0.f;
      for (int j = 0; j < 16; j++) a += ef[j] * W_ep[j * 128 + c];
      s += a;
    }
    z[idx] = s * 0.125f + b_ep[c];
  }
  __syncthreads();
  if (tid < P) {
    float s = b_path[0];
    for (int c = 0; c < 128; c++) s += z[tid * 128 + c] * W_path[c];
    lps[tid] = (path_mask[tid] == 0) ? NEGV : s;
  }
  __syncthreads();
  if (tid == 0) {
    int best = 0; float bv = lps[0];
    for (int i = 1; i < P; i++) if (lps[i] > bv) { bv = lps[i]; best = i; }
    float se_ = 0.f;
    for (int i = 0; i < P; i++) se_ += expf(lps[i] - bv);
    selP = best; logpP = lps[best] - bv - logf(se_);
  }
  __syncthreads();
  int p = selP;
  if (tid < 128) xm[tid] = z[p * 128 + tid];
  if (tid >= 128 && tid < 131) xm[tid] = path_features[p * 3 + (tid - 128)];
  __syncthreads();
  if (tid < 6) {
    float s = b_mod[tid];
    for (int i = 0; i < 131; i++) s += xm[i] * W_mod[i * 6 + tid];
    lm[tid] = (mod_masks[p * 6 + tid] == 0) ? NEGV : s;
  }
  __syncthreads();
  if (tid == 0) {
    int best = 0; float bv = lm[0];
    for (int i = 1; i < 6; i++) if (lm[i] > bv) { bv = lm[i]; best = i; }
    float se_ = 0.f;
    for (int i = 0; i < 6; i++) se_ += expf(lm[i] - bv);
    selM = best; logpM = lm[best] - bv - logf(se_);
  }
  __syncthreads();
  int ms = selM;
  if (tid < 324) sp[tid] = (tid >= 2 && tid < 322) ? path_spectrum[p * 320 + tid - 2] : 0.f;
  __syncthreads();
  if (tid < 320) {
    float acc = c2b[0];
    for (int ch = 0; ch < 8; ch++) {
      float a = c1b[ch];
      for (int t = 0; t < 5; t++) a += sp[tid + t] * c1w[ch * 5 + t];
      a = a > 0.f ? a : 0.f;
      acc += a * c2w[ch];
    }
    lf[tid] = (spec_masks[((long)p * 6 + ms) * 320 + tid] == 0) ? NEGV : acc;
  }
  __syncthreads();
  if (tid == 0) {
    int best = 0; float bv = lf[0];
    for (int i = 1; i < 320; i++) if (lf[i] > bv) { bv = lf[i]; best = i; }
    float se_ = 0.f;
    for (int i = 0; i < 320; i++) se_ += expf(lf[i] - bv);
    float logpF = lf[best] - bv - logf(se_);
    float val = 0.f;
    for (int c = 0; c < 128; c++) val += g[c] * W_val[c];
    val = val / (float)N + b_val[0];
    out[0] = (float)p;
    out[1] = (float)best;
    out[2] = (float)ms;
    out[3] = logpP + logpM + logpF;
    out[4] = val;
  }
}

extern "C" void kernel_launch(void* const* d_in, const int* in_sizes, int n_in,
                              void* d_out, int out_size, void* d_ws, size_t ws_size,
                              hipStream_t stream) {
  (void)n_in; (void)out_size; (void)ws_size;
  const float* node_feat     = (const float*)d_in[0];
  const float* edge_feat     = (const float*)d_in[1];
  const int*   edge_index    = (const int*)d_in[2];
  const int*   paths         = (const int*)d_in[3];
  const int*   path_mask     = (const int*)d_in[4];
  const float* path_features = (const float*)d_in[5];
  const int*   mod_masks     = (const int*)d_in[6];
  const int*   spec_masks    = (const int*)d_in[7];
  const float* path_spectrum = (const float*)d_in[8];
  const float* W_np  = (const float*)d_in[9];
  const float* b_np  = (const float*)d_in[10];
  const float* W_ep  = (const float*)d_in[11];
  const float* b_ep  = (const float*)d_in[12];
  const float* W_msg = (const float*)d_in[13];
  const float* b_msg = (const float*)d_in[14];
  const float* W_ih  = (const float*)d_in[15];
  const float* b_ih  = (const float*)d_in[16];
  const float* W_hh  = (const float*)d_in[17];
  const float* b_hh  = (const float*)d_in[18];
  const float* W_path = (const float*)d_in[19];
  const float* b_path = (const float*)d_in[20];
  const float* W_mod  = (const float*)d_in[21];
  const float* b_mod  = (const float*)d_in[22];
  const float* c1w = (const float*)d_in[23];
  const float* c1b = (const float*)d_in[24];
  const float* c2w = (const float*)d_in[25];
  const float* c2b = (const float*)d_in[26];
  const float* W_val = (const float*)d_in[27];
  const float* b_val = (const float*)d_in[28];

  const int  N = in_sizes[0] / 16;
  const long E = (long)in_sizes[1] / 16;
  const int  P = in_sizes[4];
  const int* src = edge_index;
  const int* dst = edge_index + E;

  auto alignup = [](size_t x) { return (x + 255) & ~(size_t)255; };
  char* w = (char*)d_ws;
  float* h        = (float*)w; w += alignup((size_t)N * 128 * 4);
  float* hs       = (float*)w; w += alignup((size_t)N * 128 * 4);
  _Float16* h16   = (_Float16*)w; w += alignup((size_t)N * 128 * 2);
  _Float16* agg16 = (_Float16*)w; w += alignup((size_t)N * 128 * 2);
  float* se       = (float*)w; w += alignup((size_t)N * 16 * 4);
  float* cnt      = (float*)w; w += alignup((size_t)N * 4);
  float* g        = (float*)w; w += alignup(128 * 4);
  float* bias_vec = (float*)w; w += alignup(128 * 4);
  _Float16* Bnp = (_Float16*)w; w += alignup(128 * 32 * 2);
  _Float16* Bc  = (_Float16*)w; w += alignup(128 * 32 * 2);
  _Float16* Btm = (_Float16*)w; w += alignup(128 * 128 * 2);
  _Float16* Bih = (_Float16*)w; w += alignup(384 * 128 * 2);
  _Float16* Bhh = (_Float16*)w; w += alignup(384 * 128 * 2);

  const unsigned rowBlocks = (unsigned)((N + 15) / 16);

  hipLaunchKernelGGL(hp_prep_kernel, dim3(1), dim3(256), 0, stream,
                     W_np, W_ep, b_ep, W_msg, b_msg, W_ih, W_hh, Bnp, Bc, Btm, Bih, Bhh, bias_vec);
  // h0 = node_feat @ W_np + b_np  (writes f32 h + f16 mirror h16)
  hipLaunchKernelGGL(hp_gemm_kernel, dim3(rowBlocks), dim3(256), 0, stream,
                     node_feat, N, 16, 32, Bnp, b_np, h, h16, 128);
  // edge-constant segment sums (once)
  {
    long nse = (long)N * 16;
    hipLaunchKernelGGL(hp_zero_kernel, dim3((unsigned)((nse + 255) / 256)), dim3(256), 0, stream, se, nse);
    hipLaunchKernelGGL(hp_zero_kernel, dim3((unsigned)((N + 255) / 256)), dim3(256), 0, stream, cnt, (long)N);
    hipLaunchKernelGGL(hp_seg_edge_kernel, dim3((unsigned)((E + 255) / 256)), dim3(256), 0, stream,
                       edge_feat, dst, E, se, cnt);
  }
  for (int it = 0; it < 3; ++it) {
    long nhs = (long)N * 128;
    hipLaunchKernelGGL(hp_zero_kernel, dim3((unsigned)((nhs + 255) / 256)), dim3(256), 0, stream, hs, nhs);
    long tot = E * 32;
    hipLaunchKernelGGL(hp_scatter_h_kernel, dim3((unsigned)((tot + 255) / 256)), dim3(256), 0, stream,
                       h, src, dst, E, hs);
    hipLaunchKernelGGL(hp_agg_kernel, dim3(rowBlocks), dim3(256), 0, stream,
                       hs, se, cnt, N, Btm, Bc, bias_vec, agg16);
    hipLaunchKernelGGL(hp_gru_kernel, dim3(rowBlocks), dim3(256), 2 * 16 * 128 * 2, stream,
                       agg16, h, h16, N, Bih, Bhh, b_ih, b_hh);
  }
  hipLaunchKernelGGL(hp_zero_kernel, dim3(1), dim3(128), 0, stream, g, (long)128);
  hipLaunchKernelGGL(hp_colsum_kernel, dim3(512), dim3(128), 0, stream, h, N, g);
  hipLaunchKernelGGL(hp_heads_kernel, dim3(1), dim3(320), 0, stream,
                     edge_feat, W_ep, b_ep, paths, path_mask, path_features, mod_masks, spec_masks,
                     path_spectrum, W_path, b_path, W_mod, b_mod, c1w, c1b, c2w, c2b,
                     W_val, b_val, g, N, P, (float*)d_out);
}
